// EPMoE_33638183862749
// MI455X (gfx1250) — compile-verified, matrix operand order checked by reading, not run on hardware
//
#include <hip/hip_runtime.h>

#define T_TOK 2048
#define K_TOP 2
#define NE    8
#define HD    1024
#define ID    1024
#define MBLK  64                               // rows per workgroup tile
#define NPAD_MAX   (T_TOK * K_TOP + NE * MBLK) // 4608 padded slots max
#define NT64_MAX   (NPAD_MAX / MBLK)           // 72 row tiles
#define LROW  72                               // LDS row stride (elements) for K=64 tile

typedef __bf16 bf16;
typedef bf16  v16bf __attribute__((ext_vector_type(16)));
typedef bf16  v8bf  __attribute__((ext_vector_type(8)));
typedef bf16  v4bf  __attribute__((ext_vector_type(4)));
typedef float v8f   __attribute__((ext_vector_type(8)));
typedef float v4f   __attribute__((ext_vector_type(4)));
typedef int   v4i_g __attribute__((vector_size(16)));   // GCC vector: matches builtin param

// ---------------- fragment load (16-bit A/B layout, wave32) ----------------
static __device__ __forceinline__ v16bf ldfrag(const bf16* p) {
  union { v16bf v; v8bf h[2]; } u;
  u.h[0] = *reinterpret_cast<const v8bf*>(p);
  u.h[1] = *reinterpret_cast<const v8bf*>(p + 16);
  return u.v;
}

static __device__ __forceinline__ v8f wmma_bf16(v16bf a, v16bf b, v8f c) {
  return __builtin_amdgcn_wmma_f32_16x16x32_bf16(
      false, a, false, b, (short)0, c, false, false);
}

// ---------------- async global -> LDS copy (16B per lane, ASYNCcnt) --------
static __device__ __forceinline__ void async_cp16(const bf16* g, bf16* l) {
#if __has_builtin(__builtin_amdgcn_global_load_async_to_lds_b128)
  __builtin_amdgcn_global_load_async_to_lds_b128(
      (__attribute__((address_space(1))) v4i_g*)g,
      (__attribute__((address_space(3))) v4i_g*)l, 0, 0);
#else
  unsigned lw = (unsigned)(unsigned long long)(__attribute__((address_space(3))) char*)l;
  unsigned long long gw = (unsigned long long)g;
  asm volatile("global_load_async_to_lds_b128 %0, %1, off"
               :: "v"(lw), "v"(gw) : "memory");
#endif
}

template <int N>
static __device__ __forceinline__ void wait_async() {
#if __has_builtin(__builtin_amdgcn_s_wait_asynccnt)
  __builtin_amdgcn_s_wait_asynccnt((short)N);
#else
  asm volatile("s_wait_asynccnt %0" :: "i"(N) : "memory");
#endif
}

// ---------------- software-pipelined 16x64x32 micro-steps ------------------
// dual-matrix (gemm1): loads stay >=2 fragments ahead of consuming WMMAs
static __device__ __forceinline__ void micro2(v16bf a, const bf16* p0, const bf16* p1,
                                              v8f* accg, v8f* accu) {
  v16bf b0 = ldfrag(p0);
  v16bf b1 = ldfrag(p0 + 16 * LROW);
  v16bf b2 = ldfrag(p0 + 32 * LROW);
  accg[0] = wmma_bf16(a, b0, accg[0]);
  v16bf b3 = ldfrag(p0 + 48 * LROW);
  accg[1] = wmma_bf16(a, b1, accg[1]);
  v16bf c0 = ldfrag(p1);
  accg[2] = wmma_bf16(a, b2, accg[2]);
  v16bf c1 = ldfrag(p1 + 16 * LROW);
  accg[3] = wmma_bf16(a, b3, accg[3]);
  v16bf c2 = ldfrag(p1 + 32 * LROW);
  accu[0] = wmma_bf16(a, c0, accu[0]);
  v16bf c3 = ldfrag(p1 + 48 * LROW);
  accu[1] = wmma_bf16(a, c1, accu[1]);
  accu[2] = wmma_bf16(a, c2, accu[2]);
  accu[3] = wmma_bf16(a, c3, accu[3]);
}

// single-matrix (gemm2)
static __device__ __forceinline__ void micro1(v16bf a, const bf16* p0, v8f* acc) {
  v16bf b0 = ldfrag(p0);
  v16bf b1 = ldfrag(p0 + 16 * LROW);
  v16bf b2 = ldfrag(p0 + 32 * LROW);
  acc[0] = wmma_bf16(a, b0, acc[0]);
  v16bf b3 = ldfrag(p0 + 48 * LROW);
  acc[1] = wmma_bf16(a, b1, acc[1]);
  acc[2] = wmma_bf16(a, b2, acc[2]);
  acc[3] = wmma_bf16(a, b3, acc[3]);
}

// ---------------- conversion kernels ----------------
__global__ void k_cvt_x(const float* __restrict__ x, bf16* __restrict__ xb) {
  size_t i = (size_t)(blockIdx.x * blockDim.x + threadIdx.x) * 4;
  v4f v = *reinterpret_cast<const v4f*>(x + i);
  *reinterpret_cast<v4bf*>(xb + i) = __builtin_convertvector(v, v4bf);
}

// dst[e][d2][d1] = (bf16) src[e][d1][d2]
__global__ void k_cvt_t(const float* __restrict__ src, bf16* __restrict__ dst,
                        int D1, int D2) {
  __shared__ float tile[32][33];
  int e  = blockIdx.z;
  const float* s = src + (size_t)e * D1 * D2;
  bf16*        d = dst + (size_t)e * D1 * D2;
  int c0 = blockIdx.x * 32;
  int r0 = blockIdx.y * 32;
  for (int i = threadIdx.y; i < 32; i += 8)
    tile[i][threadIdx.x] = s[(size_t)(r0 + i) * D2 + c0 + threadIdx.x];
  __syncthreads();
  for (int i = threadIdx.y; i < 32; i += 8)
    d[(size_t)(c0 + i) * D1 + r0 + threadIdx.x] = (bf16)tile[threadIdx.x][i];
}

// ---------------- routing ----------------
__global__ void k_count(const int* __restrict__ ids, unsigned int* __restrict__ cnt) {
  int i = blockIdx.x * blockDim.x + threadIdx.x;
  if (i < T_TOK * K_TOP) atomicAdd(&cnt[ids[i] & (NE - 1)], 1u);
}

__global__ void k_scan(const unsigned int* __restrict__ cnt,
                       unsigned int* __restrict__ off,
                       int* __restrict__ tile_expert) {
  if (blockIdx.x == 0 && threadIdx.x == 0) {
    unsigned int acc = 0;
    for (int e = 0; e < NE; ++e) {
      off[e] = acc;
      unsigned int c = (cnt[e] + (MBLK - 1u)) & ~(MBLK - 1u);
      for (unsigned int t = acc / MBLK; t < (acc + c) / MBLK; ++t) tile_expert[t] = e;
      acc += c;
    }
    off[NE] = acc;                              // n_pad
  }
}

__global__ void k_scatter(const int* __restrict__ ids, const float* __restrict__ tw,
                          const unsigned int* __restrict__ off,
                          unsigned int* __restrict__ fill,
                          int* __restrict__ slot_token, float* __restrict__ slot_w) {
  int i = blockIdx.x * blockDim.x + threadIdx.x;
  if (i < T_TOK * K_TOP) {
    int e = ids[i] & (NE - 1);
    unsigned int pos = off[e] + atomicAdd(&fill[e], 1u);
    slot_token[pos] = i >> 1;     // K_TOP == 2
    slot_w[pos] = tw[i];
  }
}

// ---------------- GEMM1: h = silu(x W0) * (x W1), bf16 out ----------------
__global__ void __launch_bounds__(256)
k_gemm1(const bf16* __restrict__ xb, const bf16* __restrict__ w0t,
        const bf16* __restrict__ w1t, const int* __restrict__ slot_token,
        const unsigned int* __restrict__ off, const int* __restrict__ tile_expert,
        const bf16* __restrict__ zrow, bf16* __restrict__ hbuf) {
  __shared__ bf16 sB0[2][128 * LROW];   // 36 KB
  __shared__ bf16 sB1[2][128 * LROW];   // 36 KB
  unsigned int npad = off[NE];
  int tile = blockIdx.x;
  if ((unsigned int)(tile * MBLK) >= npad) return;
  int e    = tile_expert[tile];
  int lane = threadIdx.x & 31;
  int wid  = threadIdx.x >> 5;
  int wm   = wid & 3;
  int wn   = wid >> 2;
  int ln   = lane & 15;
  int hi   = lane >> 4;
  int koff = hi * 8;
  int row_base = tile * MBLK + wm * 16;
  int n_blk    = blockIdx.y * 128;

  int tid   = threadIdx.x;
  int prow  = tid >> 3;             // 0..31
  int pseg  = tid & 7;              // 16B segment (0..7)
  const bf16* g0 = w0t + ((size_t)e * ID + n_blk + prow) * HD + pseg * 8;
  const bf16* g1 = w1t + ((size_t)e * ID + n_blk + prow) * HD + pseg * 8;
  int lbase = prow * LROW + pseg * 8;

  int tok = slot_token[row_base + ln];
  const bf16* arow = (tok >= 0) ? (xb + (size_t)tok * HD) : zrow;

  v8f accg[4] = {};
  v8f accu[4] = {};

#pragma unroll
  for (int j = 0; j < 4; ++j) {
    async_cp16(g0 + (size_t)(32 * j) * HD, &sB0[0][lbase + 32 * j * LROW]);
    async_cp16(g1 + (size_t)(32 * j) * HD, &sB1[0][lbase + 32 * j * LROW]);
  }

  for (int k0 = 0; k0 < HD; k0 += 64) {
    int buf = (k0 >> 6) & 1;
    // A fragments first: global latency hidden behind async-wait + barrier
    v16bf a0 = ldfrag(arow + k0 + koff);
    v16bf a1 = ldfrag(arow + k0 + 32 + koff);
    if (k0 + 64 < HD) {
      int nb = buf ^ 1;
#pragma unroll
      for (int j = 0; j < 4; ++j) {
        async_cp16(g0 + (size_t)(32 * j) * HD + k0 + 64, &sB0[nb][lbase + 32 * j * LROW]);
        async_cp16(g1 + (size_t)(32 * j) * HD + k0 + 64, &sB1[nb][lbase + 32 * j * LROW]);
      }
      wait_async<8>();               // current buffer's 8 copies done (in-order)
    } else {
      wait_async<0>();
    }
    __syncthreads();

    const bf16* p0 = &sB0[buf][(wn * 64 + ln) * LROW + koff];
    const bf16* p1 = &sB1[buf][(wn * 64 + ln) * LROW + koff];
    micro2(a0, p0,      p1,      accg, accu);
    micro2(a1, p0 + 32, p1 + 32, accg, accu);
    __syncthreads();
  }

#pragma unroll
  for (int nt = 0; nt < 4; ++nt)
#pragma unroll
    for (int r = 0; r < 8; ++r) {
      size_t slot = (size_t)row_base + r + hi * 8;
      float g = accg[nt][r];
      float u = accu[nt][r];
      float hv = (g * (1.f / (1.f + __expf(-g)))) * u;
      hbuf[slot * ID + n_blk + wn * 64 + nt * 16 + ln] = (bf16)hv;
    }
}

// ---------------- GEMM2: out[tok] += w * (h Wo) ----------------
__global__ void __launch_bounds__(256)
k_gemm2(const bf16* __restrict__ hbuf, const bf16* __restrict__ wot,
        const int* __restrict__ slot_token, const float* __restrict__ slot_w,
        const unsigned int* __restrict__ off, const int* __restrict__ tile_expert,
        float* __restrict__ out) {
  __shared__ bf16 sB[2][128 * LROW];    // 36 KB
  unsigned int npad = off[NE];
  int tile = blockIdx.x;
  if ((unsigned int)(tile * MBLK) >= npad) return;
  int e    = tile_expert[tile];
  int lane = threadIdx.x & 31;
  int wid  = threadIdx.x >> 5;
  int wm   = wid & 3;
  int wn   = wid >> 2;
  int ln   = lane & 15;
  int hi   = lane >> 4;
  int koff = hi * 8;
  int row_base = tile * MBLK + wm * 16;
  int n_blk    = blockIdx.y * 128;

  int tid  = threadIdx.x;
  int prow = tid >> 3;
  int pseg = tid & 7;
  const bf16* gb = wot + ((size_t)e * HD + n_blk + prow) * ID + pseg * 8;
  int lbase = prow * LROW + pseg * 8;

  const bf16* arow = hbuf + ((size_t)row_base + ln) * ID;

  v8f acc[4] = {};

#pragma unroll
  for (int j = 0; j < 4; ++j)
    async_cp16(gb + (size_t)(32 * j) * ID, &sB[0][lbase + 32 * j * LROW]);

  for (int k0 = 0; k0 < ID; k0 += 64) {
    int buf = (k0 >> 6) & 1;
    v16bf a0 = ldfrag(arow + k0 + koff);
    v16bf a1 = ldfrag(arow + k0 + 32 + koff);
    if (k0 + 64 < ID) {
      int nb = buf ^ 1;
#pragma unroll
      for (int j = 0; j < 4; ++j)
        async_cp16(gb + (size_t)(32 * j) * ID + k0 + 64, &sB[nb][lbase + 32 * j * LROW]);
      wait_async<4>();
    } else {
      wait_async<0>();
    }
    __syncthreads();

    const bf16* p0 = &sB[buf][(wn * 64 + ln) * LROW + koff];
    micro1(a0, p0, acc);
    micro1(a1, p0 + 32, acc);
    __syncthreads();
  }

#pragma unroll
  for (int nt = 0; nt < 4; ++nt)
#pragma unroll
    for (int r = 0; r < 8; ++r) {
      int slot = row_base + r + hi * 8;
      int tok = slot_token[slot];
      if (tok >= 0) {
        float wv = slot_w[slot];
        atomicAdd(out + (size_t)tok * HD + n_blk + wn * 64 + nt * 16 + ln,
                  wv * acc[nt][r]);
      }
    }
}

// ---------------- workspace layout ----------------
static constexpr size_t a256(size_t x) { return (x + 255) & ~(size_t)255; }
static constexpr size_t SZ_XB = (size_t)T_TOK * HD * 2;
static constexpr size_t SZ_W  = (size_t)NE * HD * ID * 2;
static constexpr size_t SZ_HB = (size_t)NPAD_MAX * ID * 2;
static constexpr size_t O_XB  = 0;
static constexpr size_t O_W0T = a256(O_XB + SZ_XB);
static constexpr size_t O_W1T = a256(O_W0T + SZ_W);
static constexpr size_t O_WOT = a256(O_W1T + SZ_W);
static constexpr size_t O_HB  = a256(O_WOT + SZ_W);
static constexpr size_t O_ST  = a256(O_HB + SZ_HB);            // int[NPAD_MAX]
static constexpr size_t O_SW  = a256(O_ST + NPAD_MAX * 4);     // float[NPAD_MAX]
static constexpr size_t O_CNT = a256(O_SW + NPAD_MAX * 4);     // uint[NE]
static constexpr size_t O_OFF = a256(O_CNT + 64);              // uint[NE+1]
static constexpr size_t O_FILL= a256(O_OFF + 64);              // uint[NE]
static constexpr size_t O_TE  = a256(O_FILL + 64);             // int[NT64_MAX]
static constexpr size_t O_ZR  = a256(O_TE + ((NT64_MAX * 4 + 255) & ~255)); // bf16[HD]

extern "C" void kernel_launch(void* const* d_in, const int* in_sizes, int n_in,
                              void* d_out, int out_size, void* d_ws, size_t ws_size,
                              hipStream_t stream) {
  const float* x  = (const float*)d_in[0];
  const float* tw = (const float*)d_in[1];
  const int*   id = (const int*)d_in[2];
  const float* wi0 = (const float*)d_in[3];
  const float* wi1 = (const float*)d_in[4];
  const float* wo  = (const float*)d_in[5];
  float* out = (float*)d_out;
  char* ws = (char*)d_ws;

  bf16* xb   = (bf16*)(ws + O_XB);
  bf16* w0t  = (bf16*)(ws + O_W0T);
  bf16* w1t  = (bf16*)(ws + O_W1T);
  bf16* wot  = (bf16*)(ws + O_WOT);
  bf16* hbuf = (bf16*)(ws + O_HB);
  int*  slot_token = (int*)(ws + O_ST);
  float* slot_w    = (float*)(ws + O_SW);
  unsigned int* cnt  = (unsigned int*)(ws + O_CNT);
  unsigned int* off  = (unsigned int*)(ws + O_OFF);
  unsigned int* fill = (unsigned int*)(ws + O_FILL);
  int*  tile_expert  = (int*)(ws + O_TE);
  bf16* zrow         = (bf16*)(ws + O_ZR);

  (void)hipMemsetAsync(ws + O_CNT, 0, O_TE - O_CNT, stream);      // cnt/off/fill
  (void)hipMemsetAsync(ws + O_ST, 0xFF, NPAD_MAX * 4, stream);    // slot_token = -1
  (void)hipMemsetAsync(ws + O_ZR, 0, HD * 2, stream);             // zero bf16 row
  (void)hipMemsetAsync(d_out, 0, (size_t)T_TOK * HD * 4, stream); // out = 0

  k_cvt_x<<<(T_TOK * HD / 4 + 255) / 256, 256, 0, stream>>>(x, xb);
  dim3 tb(32, 8);
  k_cvt_t<<<dim3(ID / 32, HD / 32, NE), tb, 0, stream>>>(wi0, w0t, HD, ID); // (H,I)->(I,H)
  k_cvt_t<<<dim3(ID / 32, HD / 32, NE), tb, 0, stream>>>(wi1, w1t, HD, ID);
  k_cvt_t<<<dim3(HD / 32, ID / 32, NE), tb, 0, stream>>>(wo, wot, ID, HD);  // (I,H)->(H,I)

  int rblocks = (T_TOK * K_TOP + 255) / 256;
  k_count<<<rblocks, 256, 0, stream>>>(id, cnt);
  k_scan<<<1, 32, 0, stream>>>(cnt, off, tile_expert);
  k_scatter<<<rblocks, 256, 0, stream>>>(id, tw, off, fill, slot_token, slot_w);

  k_gemm1<<<dim3(NT64_MAX, ID / 128), 256, 0, stream>>>(
      xb, w0t, w1t, slot_token, off, tile_expert, zrow, hbuf);
  k_gemm2<<<dim3(NT64_MAX, HD / 128), 256, 0, stream>>>(
      hbuf, wot, slot_token, slot_w, off, tile_expert, out);
}
